// IncidenceAttention_89627377533142
// MI455X (gfx1250) — compile-verified, matrix operand order checked by reading, not run on hardware
//
#include <hip/hip_runtime.h>
#include <hip/hip_bf16.h>

// ---------------------------------------------------------------------------
// IncidenceAttention for MI455X (gfx1250, wave32, WMMA bf16)
//
//   mobius(q,k) = q.swap(k)  =>  incidence = SCALE * q . (k + mw*swap(k))
//   fold mw*swap into Wk (and bk);  fold SCALE*log2(e) into Q so softmax can
//   use exp2 (v_exp_f32) directly.
//
//   GEMMs: 32(M)x64(N) tile per wave, K-step 32, explicit double buffering so
//   global_load_b128 of iteration k+1 overlaps the 8 WMMAs of iteration k.
// ---------------------------------------------------------------------------

typedef __attribute__((ext_vector_type(16))) __bf16 v16bf;
typedef __attribute__((ext_vector_type(8)))  float  v8f;

#define DIMV   1024
#define NHEADS 16
#define HDIM   64
#define SEQ    2048
#define BATCH  2
#define MTOT   (BATCH * SEQ)              // 4096 rows
#define QK_SCALE (0.125f * 1.44269504088896340736f)  // 1/sqrt(64) * log2(e)

#define WMMA_BF16(A, B, C) \
    __builtin_amdgcn_wmma_f32_16x16x32_bf16(false, (A), false, (B), (short)0, (C), false, false)

// ---- fragment loaders (ISA 7.12.2 16-bit A/B layout: lane = row/col 0..15,
// ---- VGPR0-3 = K (kh)..(kh+7), VGPR4-7 = K 16+kh..16+kh+7, kh = (lane>=16)*8)
__device__ __forceinline__ v16bf load_frag(const __bf16* __restrict__ p, int ld) {
    const int lane = threadIdx.x & 31;
    const int r    = lane & 15;
    const int kh   = (lane >> 4) << 3;
    const __bf16* q = p + (size_t)r * ld + kh;
    v16bf out;
#pragma unroll
    for (int i = 0; i < 8; ++i) { out[i] = q[i]; out[i + 8] = q[16 + i]; }
    return out;
}

__device__ __forceinline__ v16bf load_pfrag(const __bf16 p[16][48]) {
    const int lane = threadIdx.x & 31;
    const int r    = lane & 15;
    const int kh   = (lane >> 4) << 3;
    v16bf out;
#pragma unroll
    for (int i = 0; i < 8; ++i) { out[i] = p[r][kh + i]; out[i + 8] = p[r][16 + kh + i]; }
    return out;
}

// ---- width-16 row reductions (rows live across 16 lanes in C-frag layout)
__device__ __forceinline__ float rowmax16(float v) {
    v = fmaxf(v, __shfl_xor(v, 1, 16));
    v = fmaxf(v, __shfl_xor(v, 2, 16));
    v = fmaxf(v, __shfl_xor(v, 4, 16));
    v = fmaxf(v, __shfl_xor(v, 8, 16));
    return v;
}
__device__ __forceinline__ float rowsum16(float v) {
    v += __shfl_xor(v, 1, 16);
    v += __shfl_xor(v, 2, 16);
    v += __shfl_xor(v, 4, 16);
    v += __shfl_xor(v, 8, 16);
    return v;
}

// ---------------------------------------------------------------------------
// 1) x (f32) -> bf16
__global__ __launch_bounds__(256) void cvt_x_kernel(const float* __restrict__ x,
                                                    __bf16* __restrict__ xb, int n) {
    int i = (blockIdx.x * 256 + threadIdx.x) * 4;
    if (i + 3 < n) {
        float4 f = *(const float4*)(x + i);
        xb[i + 0] = (__bf16)f.x; xb[i + 1] = (__bf16)f.y;
        xb[i + 2] = (__bf16)f.z; xb[i + 3] = (__bf16)f.w;
    }
}

// 2) weights -> bf16; for Wk fold the mobius swap: Wk'[j,:] = Wk[j,:] + mw*Wk[swap(j),:]
__global__ __launch_bounds__(256) void cvt_w_kernel(const float* __restrict__ Wq,
                                                    const float* __restrict__ Wk,
                                                    const float* __restrict__ Wv,
                                                    const float* __restrict__ Wo,
                                                    const float* __restrict__ mw,
                                                    __bf16* __restrict__ out) {
    const int z = blockIdx.y;  // 0:Wq 1:Wk 2:Wv 3:Wo
    const float* W = (z == 0) ? Wq : (z == 1) ? Wk : (z == 2) ? Wv : Wo;
    __bf16* o = out + (size_t)z * DIMV * DIMV;
    int i = (blockIdx.x * 256 + threadIdx.x) * 4;
    if (z == 1) {
        const int   row = i >> 10, col = i & 1023;
        const int   d   = row & 63;
        const int   rp  = (row & ~63) | ((d + 32) & 63);
        const float m   = mw[0];
#pragma unroll
        for (int t = 0; t < 4; ++t)
            o[i + t] = (__bf16)(W[i + t] + m * W[(size_t)rp * DIMV + col + t]);
    } else {
#pragma unroll
        for (int t = 0; t < 4; ++t) o[i + t] = (__bf16)W[i + t];
    }
}

// ---- 32(M) x 64(N) per-wave GEMM mainloop, double-buffered:
// ----   C = X(M,K) * W(N,K)^T ;  c[i*4+j] = rows m0+i*16, cols n0+j*16
__device__ __forceinline__ void gemm_tile32(const __bf16* __restrict__ X,
                                            const __bf16* __restrict__ W,
                                            int m0, int n0, v8f c[8]) {
    const v8f zero = {0.f, 0.f, 0.f, 0.f, 0.f, 0.f, 0.f, 0.f};
#pragma unroll
    for (int t = 0; t < 8; ++t) c[t] = zero;

    const __bf16* Xp = X + (size_t)m0 * DIMV;
    const __bf16* Wp = W + (size_t)n0 * DIMV;

    v16bf a0 = load_frag(Xp, DIMV);
    v16bf a1 = load_frag(Xp + (size_t)16 * DIMV, DIMV);
    v16bf b[4];
#pragma unroll
    for (int j = 0; j < 4; ++j) b[j] = load_frag(Wp + (size_t)(j * 16) * DIMV, DIMV);

#pragma unroll 2
    for (int k = 0; k < DIMV; k += 32) {
        // issue next iteration's fragment loads (clamped, branchless tail)
        const int kn = (k + 32 < DIMV) ? (k + 32) : 0;
        v16bf a0n = load_frag(Xp + kn, DIMV);
        v16bf a1n = load_frag(Xp + (size_t)16 * DIMV + kn, DIMV);
        v16bf bn[4];
#pragma unroll
        for (int j = 0; j < 4; ++j) bn[j] = load_frag(Wp + (size_t)(j * 16) * DIMV + kn, DIMV);

        // 8 WMMAs on the resident fragments
#pragma unroll
        for (int j = 0; j < 4; ++j) {
            c[j]     = WMMA_BF16(a0, b[j], c[j]);
            c[4 + j] = WMMA_BF16(a1, b[j], c[4 + j]);
        }
        a0 = a0n; a1 = a1n;
#pragma unroll
        for (int j = 0; j < 4; ++j) b[j] = bn[j];
    }
}

// 3) QKV projections.  z=0: Q (scaled by QK_SCALE), z=1: K', z=2: V (stored transposed)
__global__ __launch_bounds__(128) void qkv_kernel(const __bf16* __restrict__ Xb,
                                                  const __bf16* __restrict__ Wb,
                                                  const float* __restrict__ bq,
                                                  const float* __restrict__ bk,
                                                  const float* __restrict__ bv,
                                                  const float* __restrict__ mw,
                                                  __bf16* __restrict__ Qb,
                                                  __bf16* __restrict__ Kpb,
                                                  __bf16* __restrict__ Vtb) {
    const int z    = blockIdx.z;
    const int wave = threadIdx.x >> 5;
    const int m0   = blockIdx.x * 128 + wave * 32;
    const int n0   = blockIdx.y * 64;

    v8f c[8];
    gemm_tile32(Xb, Wb + (size_t)z * DIMV * DIMV, m0, n0, c);

    const int lane  = threadIdx.x & 31;
    const int colb  = lane & 15;
    const int rbase = (lane >> 4) << 3;
    const float mwv = (z == 1) ? mw[0] : 0.f;

#pragma unroll
    for (int j = 0; j < 4; ++j) {
        const int n = n0 + j * 16 + colb;
        const int h = n >> 6, d = n & 63;
        float bias;
        if (z == 0)      bias = bq[n];
        else if (z == 1) bias = bk[n] + mwv * bk[(n & ~63) | ((d + 32) & 63)];
        else             bias = bv[n];
#pragma unroll
        for (int i = 0; i < 2; ++i) {
#pragma unroll
            for (int g = 0; g < 8; ++g) {
                const int row = m0 + i * 16 + rbase + g;  // 0..4095
                const int b   = row >> 11, s = row & 2047;
                const int bh  = b * NHEADS + h;
                float v = c[i * 4 + j][g] + bias;
                if (z == 0) {
                    v *= QK_SCALE;
                    Qb[((size_t)bh * SEQ + s) * HDIM + d] = (__bf16)v;
                } else if (z == 1) {
                    Kpb[((size_t)bh * SEQ + s) * HDIM + d] = (__bf16)v;
                } else {
                    Vtb[((size_t)bh * HDIM + d) * SEQ + s] = (__bf16)v;  // transposed
                }
            }
        }
    }
}

// 4) flash attention: softmax(Q K'^T) V, per (b,h); 4 waves/block, 16 queries/wave
__global__ __launch_bounds__(128) void attn_kernel(const __bf16* __restrict__ Qb,
                                                   const __bf16* __restrict__ Kpb,
                                                   const __bf16* __restrict__ Vtb,
                                                   __bf16* __restrict__ Ab) {
    __shared__ __bf16 pbuf[4][16][48];   // per-wave P tile (16x32, padded rows)

    const int wave = threadIdx.x >> 5;
    const int lane = threadIdx.x & 31;
    const int bh   = blockIdx.y;
    const int b    = bh >> 4, h = bh & 15;
    const int q0   = blockIdx.x * 64 + wave * 16;

    const __bf16* Qs  = Qb  + (size_t)bh * SEQ * HDIM;
    const __bf16* Ks  = Kpb + (size_t)bh * SEQ * HDIM;
    const __bf16* Vts = Vtb + (size_t)bh * HDIM * SEQ;

    // stationary Q fragments (d 0..31 and 32..63)
    const v16bf aQ0 = load_frag(Qs + (size_t)q0 * HDIM + 0,  HDIM);
    const v16bf aQ1 = load_frag(Qs + (size_t)q0 * HDIM + 32, HDIM);

    const v8f zero = {0.f, 0.f, 0.f, 0.f, 0.f, 0.f, 0.f, 0.f};
    v8f c[4];
#pragma unroll
    for (int j = 0; j < 4; ++j) c[j] = zero;
    float mrun[8], lrun[8];
#pragma unroll
    for (int g = 0; g < 8; ++g) { mrun[g] = -3.0e38f; lrun[g] = 0.f; }

    const int colb  = lane & 15;
    const int rbase = (lane >> 4) << 3;

    // preload K' fragments for chunk 0
    v16bf kf[4];
    kf[0] = load_frag(Ks + 0 * HDIM + 0,  HDIM);
    kf[1] = load_frag(Ks + 0 * HDIM + 32, HDIM);
    kf[2] = load_frag(Ks + 16 * HDIM + 0,  HDIM);
    kf[3] = load_frag(Ks + 16 * HDIM + 32, HDIM);

#pragma unroll 1
    for (int key0 = 0; key0 < SEQ; key0 += 32) {
        // S = Q K'^T for 32 keys using resident fragments
        v8f s0 = zero, s1 = zero;
        s0 = WMMA_BF16(aQ0, kf[0], s0);
        s0 = WMMA_BF16(aQ1, kf[1], s0);
        s1 = WMMA_BF16(aQ0, kf[2], s1);
        s1 = WMMA_BF16(aQ1, kf[3], s1);

        // issue V loads for this chunk + K' loads for the next chunk NOW, so
        // they overlap the softmax VALU block below (branchless clamped tail)
        v16bf bV[4];
#pragma unroll
        for (int j = 0; j < 4; ++j)
            bV[j] = load_frag(Vts + (size_t)(j * 16) * SEQ + key0, SEQ);
        const int kn = (key0 + 32 < SEQ) ? (key0 + 32) : 0;
        v16bf kfn[4];
        kfn[0] = load_frag(Ks + (size_t)kn * HDIM + 0,  HDIM);
        kfn[1] = load_frag(Ks + (size_t)kn * HDIM + 32, HDIM);
        kfn[2] = load_frag(Ks + (size_t)(kn + 16) * HDIM + 0,  HDIM);
        kfn[3] = load_frag(Ks + (size_t)(kn + 16) * HDIM + 32, HDIM);
        __builtin_prefetch(Vts + (size_t)kn, 0, 0);  // next V chunk -> cache

        // online softmax update (scale already folded into Q; exp2 domain)
        float alph[8];
#pragma unroll
        for (int g = 0; g < 8; ++g) {
            float r  = rowmax16(fmaxf(s0[g], s1[g]));
            float mn = fmaxf(mrun[g], r);
            alph[g]  = exp2f(mrun[g] - mn);
            mrun[g]  = mn;
            float e0 = exp2f(s0[g] - mn);
            float e1 = exp2f(s1[g] - mn);
            lrun[g]  = lrun[g] * alph[g] + rowsum16(e0 + e1);
            s0[g] = e0; s1[g] = e1;
        }
#pragma unroll
        for (int j = 0; j < 4; ++j)
#pragma unroll
            for (int g = 0; g < 8; ++g) c[j][g] *= alph[g];

        // C-frag layout -> A-frag layout for P via per-wave LDS tile
#pragma unroll
        for (int g = 0; g < 8; ++g) {
            const int row = rbase + g;
            pbuf[wave][row][colb]      = (__bf16)s0[g];
            pbuf[wave][row][16 + colb] = (__bf16)s1[g];
        }
        asm volatile("s_wait_dscnt 0" ::: "memory");
        const v16bf aP = load_pfrag(pbuf[wave]);

        // O += P * V   (B frags from transposed V: contiguous along keys)
#pragma unroll
        for (int j = 0; j < 4; ++j) c[j] = WMMA_BF16(aP, bV[j], c[j]);

#pragma unroll
        for (int j = 0; j < 4; ++j) kf[j] = kfn[j];
    }

    // normalize and store merged-head bf16 activations for the output GEMM
    float inv[8];
#pragma unroll
    for (int g = 0; g < 8; ++g) inv[g] = 1.0f / lrun[g];
#pragma unroll
    for (int j = 0; j < 4; ++j) {
        const int d = j * 16 + colb;
#pragma unroll
        for (int g = 0; g < 8; ++g) {
            const int s = q0 + rbase + g;
            Ab[((size_t)(b * SEQ + s)) * DIMV + h * HDIM + d] = (__bf16)(c[j][g] * inv[g]);
        }
    }
}

// 5) output projection: out = attn @ Wo^T + bo   (fp32 result)
__global__ __launch_bounds__(128) void oproj_kernel(const __bf16* __restrict__ Ab,
                                                    const __bf16* __restrict__ Wob,
                                                    const float* __restrict__ bo,
                                                    float* __restrict__ out) {
    const int wave = threadIdx.x >> 5;
    const int m0   = blockIdx.x * 128 + wave * 32;
    const int n0   = blockIdx.y * 64;

    v8f c[8];
    gemm_tile32(Ab, Wob, m0, n0, c);

    const int lane  = threadIdx.x & 31;
    const int colb  = lane & 15;
    const int rbase = (lane >> 4) << 3;
#pragma unroll
    for (int j = 0; j < 4; ++j) {
        const int n = n0 + j * 16 + colb;
        const float bias = bo[n];
#pragma unroll
        for (int i = 0; i < 2; ++i) {
#pragma unroll
            for (int g = 0; g < 8; ++g) {
                const int row = m0 + i * 16 + rbase + g;
                out[(size_t)row * DIMV + n] = c[i * 4 + j][g] + bias;
            }
        }
    }
}

// ---------------------------------------------------------------------------
extern "C" void kernel_launch(void* const* d_in, const int* in_sizes, int n_in,
                              void* d_out, int out_size, void* d_ws, size_t ws_size,
                              hipStream_t stream) {
    const float* x  = (const float*)d_in[0];
    const float* Wq = (const float*)d_in[1];
    const float* bq = (const float*)d_in[2];
    const float* Wk = (const float*)d_in[3];
    const float* bk = (const float*)d_in[4];
    const float* Wv = (const float*)d_in[5];
    const float* bv = (const float*)d_in[6];
    const float* Wo = (const float*)d_in[7];
    const float* bo = (const float*)d_in[8];
    const float* mw = (const float*)d_in[9];
    float* out = (float*)d_out;

    char* ws = (char*)d_ws;
    __bf16* Xb  = (__bf16*)(ws);                        //  8 MB  x in bf16
    __bf16* Wb  = (__bf16*)(ws + ((size_t)8  << 20));   //  8 MB  4 weight mats bf16
    __bf16* Qb  = (__bf16*)(ws + ((size_t)16 << 20));   //  8 MB  Q  [bh][s][d]
    __bf16* Kpb = (__bf16*)(ws + ((size_t)24 << 20));   //  8 MB  K' [bh][s][d]
    __bf16* Vtb = (__bf16*)(ws + ((size_t)32 << 20));   //  8 MB  V^T[bh][d][s]
    __bf16* Ab  = (__bf16*)(ws + ((size_t)40 << 20));   //  8 MB  attn out bf16

    const int nX = MTOT * DIMV;                         // 4M elements
    cvt_x_kernel<<<dim3(nX / (256 * 4)), dim3(256), 0, stream>>>(x, Xb, nX);
    cvt_w_kernel<<<dim3((DIMV * DIMV) / (256 * 4), 4), dim3(256), 0, stream>>>(
        Wq, Wk, Wv, Wo, mw, Wb);

    qkv_kernel<<<dim3(MTOT / 128, DIMV / 64, 3), dim3(128), 0, stream>>>(
        Xb, Wb, bq, bk, bv, mw, Qb, Kpb, Vtb);

    attn_kernel<<<dim3(SEQ / 64, BATCH * NHEADS), dim3(128), 0, stream>>>(
        Qb, Kpb, Vtb, Ab);

    oproj_kernel<<<dim3(MTOT / 128, DIMV / 64), dim3(128), 0, stream>>>(
        Ab, Wb + (size_t)3 * DIMV * DIMV, bo, out);

    (void)in_sizes; (void)n_in; (void)out_size; (void)ws_size;
}
